// PNA_GNN_6408091205938
// MI455X (gfx1250) — matrix operand order repaired and checked by reading, not compiled, not run on hardware
//
#include <hip/hip_runtime.h>
#include <hip/hip_bf16.h>
#include <math.h>

typedef __bf16 bf16_t;
typedef __attribute__((ext_vector_type(16))) __bf16 v16bf;
typedef __attribute__((ext_vector_type(8)))  __bf16 v8bf;
typedef __attribute__((ext_vector_type(8)))  float  v8f;

union AFrag { v16bf v; bf16_t e[16]; };
union CFrag { v8f  v; float  e[8];  };

// Pack a 16x32 bf16 A-fragment row slice in contiguous k order (B loads use the same):
// lanes 0-15 hold K=[0..15], lanes 16-31 hold K=[16..31]
// for row M = lane%16. rowk points at the K-tile base of this lane's row.
__device__ __forceinline__ v16bf pack_a(const bf16_t* rowk, int half) {
  AFrag f;
  v8bf lo = *(const v8bf*)(rowk + half * 16);
  v8bf hi = *(const v8bf*)(rowk + half * 16 + 8);
#pragma unroll
  for (int i = 0; i < 8; ++i) { f.e[i] = lo[i]; f.e[8 + i] = hi[i]; }
  return f.v;
}

// ---------------- CSR / degree / avg_log ----------------

__global__ void zero_u32_kernel(unsigned* __restrict__ p, int n) {
  int i = blockIdx.x * blockDim.x + threadIdx.x;
  if (i < n) p[i] = 0u;
}

__global__ void deg_kernel(const int* __restrict__ ei, unsigned* __restrict__ cnt, int E) {
  int e = blockIdx.x * blockDim.x + threadIdx.x;
  if (e < E) atomicAdd(&cnt[ei[E + e]], 1u);
}

__global__ void scan_kernel(const unsigned* __restrict__ cnt, unsigned* __restrict__ rowptr,
                            unsigned* __restrict__ cursor, float* __restrict__ logsum,
                            int N, int E) {
  __shared__ unsigned part[1024];
  __shared__ float    ls[1024];
  int tid = threadIdx.x;
  int chunk = (N + 1023) >> 10;
  int b = tid * chunk;
  int e = b + chunk; if (e > N) e = N;
  unsigned s = 0; float l = 0.f;
  for (int i = b; i < e; ++i) { s += cnt[i]; l += logf((float)cnt[i] + 1.0f); }
  part[tid] = s; ls[tid] = l;
  __syncthreads();
  for (int off = 1; off < 1024; off <<= 1) {
    unsigned v = (tid >= off) ? part[tid - off] : 0u;
    __syncthreads();
    part[tid] += v;
    __syncthreads();
  }
  unsigned run = part[tid] - s;  // exclusive prefix for this chunk
  for (int i = b; i < e; ++i) { rowptr[i] = run; cursor[i] = run; run += cnt[i]; }
  if (tid == 0) rowptr[N] = (unsigned)E;
  __syncthreads();
  for (int o = 512; o > 0; o >>= 1) { if (tid < o) ls[tid] += ls[tid + o]; __syncthreads(); }
  if (tid == 0) logsum[0] = ls[0];
}

__global__ void scatter_kernel(const int* __restrict__ ei, unsigned* __restrict__ cursor,
                               unsigned* __restrict__ perm, int E) {
  int e = blockIdx.x * blockDim.x + threadIdx.x;
  if (e >= E) return;
  unsigned pos = atomicAdd(&cursor[ei[E + e]], 1u);
  perm[pos] = (unsigned)e;
}

__global__ void cvt_bf16_kernel(const float* __restrict__ x, bf16_t* __restrict__ y, int n) {
  int i = blockIdx.x * blockDim.x + threadIdx.x;
  if (i < n) y[i] = (bf16_t)x[i];
}

// ---------------- weight folding ----------------
// W_eff = we @ wp[2F:3F,:]  (13xF, padded to 32xF), b_m = be @ wp[2F:3F,:] + bp
// WpTd/WpTs: bf16 transposed [F][F] views of wp[0:F,:] / wp[F:2F,:]
__global__ void fold_edge_kernel(const float* __restrict__ we, const float* __restrict__ be,
                                 const float* __restrict__ wp, const float* __restrict__ bp,
                                 bf16_t* __restrict__ WpTd, bf16_t* __restrict__ WpTs,
                                 bf16_t* __restrict__ WeT, float* __restrict__ bm, int F) {
  int idx = blockIdx.x * blockDim.x + threadIdx.x;
  int FF = F * F;
  if (idx < FF) {
    int k = idx / F, f = idx - k * F;
    WpTd[(size_t)f * F + k] = (bf16_t)wp[(size_t)k * F + f];
  } else if (idx < 2 * FF) {
    int r = idx - FF; int k = r / F, f = r - k * F;
    WpTs[(size_t)f * F + k] = (bf16_t)wp[(size_t)(F + k) * F + f];
  } else if (idx < 2 * FF + 32 * F) {
    int r = idx - 2 * FF; int f = r >> 5, a = r & 31;
    float v = 0.0f;
    if (a < 13)
      for (int j = 0; j < F; ++j) v += we[a * F + j] * wp[(size_t)(2 * F + j) * F + f];
    WeT[f * 32 + a] = (bf16_t)v;
  } else if (idx < 2 * FF + 32 * F + F) {
    int f = idx - 2 * FF - 32 * F;
    float v = bp[f];
    for (int j = 0; j < F; ++j) v += be[j] * wp[(size_t)(2 * F + j) * F + f];
    bm[f] = v;
  }
}

// W2 = wo @ wl  stored transposed bf16 [Fo][13F]; b2 = bo @ wl + bl
__global__ void fold_node_kernel(const float* __restrict__ wo, const float* __restrict__ bo,
                                 const float* __restrict__ wl, const float* __restrict__ bl,
                                 bf16_t* __restrict__ W2T, float* __restrict__ b2,
                                 int K13, int Fo) {
  int idx = blockIdx.x * blockDim.x + threadIdx.x;
  if (idx < K13 * Fo) {
    int k = idx / Fo, o = idx - k * Fo;
    float v = 0.0f;
    for (int j = 0; j < Fo; ++j) v += wo[(size_t)k * Fo + j] * wl[(size_t)j * Fo + o];
    W2T[(size_t)o * K13 + k] = (bf16_t)v;
  } else if (idx < K13 * Fo + Fo) {
    int o = idx - K13 * Fo;
    float v = bl[o];
    for (int j = 0; j < Fo; ++j) v += bo[j] * wl[(size_t)j * Fo + o];
    b2[o] = v;
  }
}

// ---------------- edge message GEMM (WMMA, NT-wide register blocking) ----------------
// m[e,:] = x[dst]@Wp_d + x[src]@Wp_s + ea_pad@W_eff + bm   (bf16 in, f32 acc, bf16 out)
// Each wave computes a 16 x (NT*16) output tile: A fragment loaded once per K-tile,
// NT WMMAs issued against NT B fragments -> NT-fold reduction of A-side traffic.
template <int NT>
__global__ void edge_msg_kernel(const int* __restrict__ ei, const float* __restrict__ ea,
                                const bf16_t* __restrict__ xbf,
                                const bf16_t* __restrict__ WpTd, const bf16_t* __restrict__ WpTs,
                                const bf16_t* __restrict__ WeT, const float* __restrict__ bm,
                                bf16_t* __restrict__ m, int E, int F) {
  const int colGroups = (F >> 4) / NT;
  const int rowTiles = E >> 4;  // E divisible by 16
  int wave = blockIdx.x * (blockDim.x >> 5) + (threadIdx.x >> 5);
  if (wave >= rowTiles * colGroups) return;  // wave-uniform: EXEC stays all-ones for WMMA
  int tr = wave / colGroups, tg = wave - tr * colGroups;
  int lane = threadIdx.x & 31;
  int r16 = lane & 15, half = lane >> 4;
  int e = tr * 16 + r16;
  int s = ei[e];
  int d = ei[E + e];
  int colb = tg * (NT * 16);

  CFrag acc[NT];
#pragma unroll
  for (int t = 0; t < NT; ++t)
#pragma unroll
    for (int i = 0; i < 8; ++i) acc[t].e[i] = 0.0f;

  const bf16_t* xd = xbf + (size_t)d * F;
  const bf16_t* xs = xbf + (size_t)s * F;
  const bf16_t* bd[NT];
  const bf16_t* bs[NT];
#pragma unroll
  for (int t = 0; t < NT; ++t) {
    bd[t] = WpTd + (size_t)(colb + t * 16 + r16) * F;
    bs[t] = WpTs + (size_t)(colb + t * 16 + r16) * F;
  }

  for (int k0 = 0; k0 < F; k0 += 32) {
    v16bf a = pack_a(xd + k0, half);
#pragma unroll
    for (int t = 0; t < NT; ++t) {
      v16bf b = *(const v16bf*)(bd[t] + k0 + half * 16);
      acc[t].v = __builtin_amdgcn_wmma_f32_16x16x32_bf16(false, a, false, b, (short)0, acc[t].v, false, false);
    }
  }
  for (int k0 = 0; k0 < F; k0 += 32) {
    v16bf a = pack_a(xs + k0, half);
#pragma unroll
    for (int t = 0; t < NT; ++t) {
      v16bf b = *(const v16bf*)(bs[t] + k0 + half * 16);
      acc[t].v = __builtin_amdgcn_wmma_f32_16x16x32_bf16(false, a, false, b, (short)0, acc[t].v, false, false);
    }
  }
  {  // folded edge-encoder part: one K=32 tile (13 real + 19 zero)
    AFrag fa;
#pragma unroll
    for (int j = 0; j < 16; ++j) {
      int k = half * 16 + j;
      fa.e[j] = (k < 13) ? (bf16_t)ea[(size_t)e * 13 + k] : (bf16_t)0.0f;
    }
#pragma unroll
    for (int t = 0; t < NT; ++t) {
      v16bf b = *(const v16bf*)(WeT + (colb + t * 16 + r16) * 32 + half * 16);
      acc[t].v = __builtin_amdgcn_wmma_f32_16x16x32_bf16(false, fa.v, false, b, (short)0, acc[t].v, false, false);
    }
  }
#pragma unroll
  for (int t = 0; t < NT; ++t) {
    int coln = colb + t * 16 + r16;
    float bias = bm[coln];
#pragma unroll
    for (int i = 0; i < 8; ++i) {
      int er = tr * 16 + half * 8 + i;  // C/D layout: M = 8*half + i, N = lane%16
      m[(size_t)er * F + coln] = (bf16_t)(acc[t].e[i] + bias);
    }
  }
}

// ---------------- per-node segment aggregation -> z row ----------------
__global__ void aggregate_kernel(const unsigned* __restrict__ rowptr, const unsigned* __restrict__ perm,
                                 const bf16_t* __restrict__ m, const float* __restrict__ xf,
                                 const unsigned* __restrict__ cnt, const float* __restrict__ logsum,
                                 bf16_t* __restrict__ z, int N, int F) {
  int wave = blockIdx.x * (blockDim.x >> 5) + (threadIdx.x >> 5);
  if (wave >= N) return;
  int n = wave;
  int lane = threadIdx.x & 31;
  unsigned beg = rowptr[n], end = rowptr[n + 1];
  float sum[4], ss[4], mn[4], mx[4];
#pragma unroll
  for (int i = 0; i < 4; ++i) { sum[i] = 0.f; ss[i] = 0.f; mn[i] = 3.0e38f; mx[i] = -3.0e38f; }
  const int ncols = F >> 5;
  for (unsigned j = beg; j < end; ++j) {
    const bf16_t* mr = m + (size_t)perm[j] * F;
    if (j + 1 < end) __builtin_prefetch(m + (size_t)perm[j + 1] * F, 0, 0);
#pragma unroll
    for (int c = 0; c < 4; ++c) {
      if (c < ncols) {
        float v = (float)mr[lane + (c << 5)];
        sum[c] += v; ss[c] += v * v;
        mn[c] = fminf(mn[c], v); mx[c] = fmaxf(mx[c], v);
      }
    }
  }
  float cf = (float)cnt[n];
  float denom = fmaxf(cf, 1.0f);
  float avg_log = logsum[0] / (float)N;
  float logd = logf(denom + 1.0f);
  float s1 = logd / avg_log;
  float s2 = avg_log / logd;
  bool has = cf > 0.0f;
  bf16_t* zr = z + (size_t)n * 13 * F;
#pragma unroll
  for (int c = 0; c < 4; ++c) {
    if (c < ncols) {
      int col = lane + (c << 5);
      float mean  = sum[c] / denom;
      float mean2 = ss[c] / denom;
      float sd  = sqrtf(fmaxf(mean2 - mean * mean, 0.0f) + 1e-5f);
      float vmn = has ? mn[c] : 0.0f;
      float vmx = has ? mx[c] : 0.0f;
      zr[col]            = (bf16_t)xf[(size_t)n * F + col];
      zr[1 * F + col]    = (bf16_t)mean;
      zr[2 * F + col]    = (bf16_t)vmn;
      zr[3 * F + col]    = (bf16_t)vmx;
      zr[4 * F + col]    = (bf16_t)sd;
      zr[5 * F + col]    = (bf16_t)(mean * s1);
      zr[6 * F + col]    = (bf16_t)(vmn * s1);
      zr[7 * F + col]    = (bf16_t)(vmx * s1);
      zr[8 * F + col]    = (bf16_t)(sd * s1);
      zr[9 * F + col]    = (bf16_t)(mean * s2);
      zr[10 * F + col]   = (bf16_t)(vmn * s2);
      zr[11 * F + col]   = (bf16_t)(vmx * s2);
      zr[12 * F + col]   = (bf16_t)(sd * s2);
    }
  }
}

// ---------------- node GEMM: Y = z @ W2^T + b2 (WMMA, NT-wide) ----------------
template <int NT>
__global__ void node_gemm_kernel(const bf16_t* __restrict__ A, const bf16_t* __restrict__ BT,
                                 const float* __restrict__ bias, float* __restrict__ Y,
                                 int Nrows, int K, int Ncols) {
  const int colGroups = (Ncols >> 4) / NT;
  const int rowTiles = Nrows >> 4;  // N divisible by 16
  int wave = blockIdx.x * (blockDim.x >> 5) + (threadIdx.x >> 5);
  if (wave >= rowTiles * colGroups) return;
  int tr = wave / colGroups, tg = wave - tr * colGroups;
  int lane = threadIdx.x & 31;
  int r16 = lane & 15, half = lane >> 4;
  int colb = tg * (NT * 16);
  const bf16_t* ar = A + (size_t)(tr * 16 + r16) * K;
  const bf16_t* br[NT];
#pragma unroll
  for (int t = 0; t < NT; ++t) br[t] = BT + (size_t)(colb + t * 16 + r16) * K;
  CFrag acc[NT];
#pragma unroll
  for (int t = 0; t < NT; ++t)
#pragma unroll
    for (int i = 0; i < 8; ++i) acc[t].e[i] = 0.0f;
  for (int k0 = 0; k0 < K; k0 += 32) {
    v16bf a = pack_a(ar + k0, half);
#pragma unroll
    for (int t = 0; t < NT; ++t) {
      v16bf b = *(const v16bf*)(br[t] + k0 + half * 16);
      acc[t].v = __builtin_amdgcn_wmma_f32_16x16x32_bf16(false, a, false, b, (short)0, acc[t].v, false, false);
    }
  }
#pragma unroll
  for (int t = 0; t < NT; ++t) {
    int coln = colb + t * 16 + r16;
    float bv = bias[coln];
#pragma unroll
    for (int i = 0; i < 8; ++i)
      Y[(size_t)(tr * 16 + half * 8 + i) * Ncols + coln] = acc[t].e[i] + bv;
  }
}

// ---------------- BatchNorm (training-mode) + ReLU ----------------
__global__ void bn_stats_kernel(const float* __restrict__ y, float* __restrict__ bsum,
                                float* __restrict__ bss, int N, int Fo) {
  __shared__ float s1[256], s2[256];
  int col = blockIdx.x;
  int tid = threadIdx.x;
  float a = 0.f, b = 0.f;
  for (int n = tid; n < N; n += blockDim.x) {
    float v = y[(size_t)n * Fo + col];
    a += v; b += v * v;
  }
  s1[tid] = a; s2[tid] = b;
  __syncthreads();
  for (int o = blockDim.x >> 1; o > 0; o >>= 1) {
    if (tid < o) { s1[tid] += s1[tid + o]; s2[tid] += s2[tid + o]; }
    __syncthreads();
  }
  if (tid == 0) { bsum[col] = s1[0]; bss[col] = s2[0]; }
}

__global__ void bn_apply_kernel(float* __restrict__ y, const float* __restrict__ bsum,
                                const float* __restrict__ bss, const float* __restrict__ g,
                                const float* __restrict__ bb, bf16_t* __restrict__ xbf,
                                int N, int Fo) {
  int idx = blockIdx.x * blockDim.x + threadIdx.x;
  if (idx >= N * Fo) return;
  int col = idx % Fo;
  float mu  = bsum[col] / (float)N;
  float var = bss[col] / (float)N - mu * mu;
  float v = g[col] * (y[idx] - mu) * rsqrtf(var + 1e-5f) + bb[col];
  v = fmaxf(v, 0.0f);
  y[idx] = v;
  xbf[idx] = (bf16_t)v;
}

// ---------------- classifier ----------------
__global__ void classify_kernel(const float* __restrict__ x, const float* __restrict__ wc,
                                const float* __restrict__ bc, float* __restrict__ out,
                                int N, int K, int C) {
  int idx = blockIdx.x * blockDim.x + threadIdx.x;
  if (idx >= N * C) return;
  int n = idx / C, c = idx - n * C;
  float s = bc[c];
  const float* xr = x + (size_t)n * K;
  for (int k = 0; k < K; ++k) s += xr[k] * wc[(size_t)k * C + c];
  out[idx] = s;
}

extern "C" void kernel_launch(void* const* d_in, const int* in_sizes, int n_in,
                              void* d_out, int out_size, void* d_ws, size_t ws_size,
                              hipStream_t stream) {
  (void)in_sizes; (void)n_in; (void)out_size; (void)ws_size;
  const int N = 50000, E = 800000, F0 = 32, C = 10;

  const float* x0 = (const float*)d_in[0];
  const int*   ei = (const int*)d_in[1];
  const float* ea = (const float*)d_in[2];
  const float* wc = (const float*)d_in[33];
  const float* bc = (const float*)d_in[34];

  uintptr_t base = (uintptr_t)d_ws;
  auto carve = [&](size_t bytes) -> void* {
    uintptr_t p = base;
    base += (bytes + 255) & ~(size_t)255;
    return (void*)p;
  };

  unsigned* cnt    = (unsigned*)carve(sizeof(unsigned) * N);
  unsigned* cursor = (unsigned*)carve(sizeof(unsigned) * N);
  unsigned* rowptr = (unsigned*)carve(sizeof(unsigned) * (N + 1));
  unsigned* perm   = (unsigned*)carve(sizeof(unsigned) * E);
  float*    logsum = (float*)carve(256);
  bf16_t*   xbf    = (bf16_t*)carve(sizeof(bf16_t) * (size_t)N * 256);
  float*    xfA    = (float*)carve(sizeof(float) * (size_t)N * 256);
  float*    xfB    = (float*)carve(sizeof(float) * (size_t)N * 256);
  bf16_t*   mbuf   = (bf16_t*)carve(sizeof(bf16_t) * (size_t)E * 128);
  bf16_t*   zbuf   = (bf16_t*)carve(sizeof(bf16_t) * (size_t)N * 13 * 128);
  float*    bnsum  = (float*)carve(sizeof(float) * 256);
  float*    bnss   = (float*)carve(sizeof(float) * 256);
  bf16_t*   WpTd   = (bf16_t*)carve(sizeof(bf16_t) * 128 * 128);
  bf16_t*   WpTs   = (bf16_t*)carve(sizeof(bf16_t) * 128 * 128);
  bf16_t*   WeT    = (bf16_t*)carve(sizeof(bf16_t) * 128 * 32);
  float*    bm     = (float*)carve(sizeof(float) * 128);
  bf16_t*   W2T    = (bf16_t*)carve(sizeof(bf16_t) * (size_t)256 * 13 * 128);
  float*    b2     = (float*)carve(sizeof(float) * 256);

  // degree histogram -> CSR -> avg_log
  zero_u32_kernel<<<(N + 255) / 256, 256, 0, stream>>>(cnt, N);
  deg_kernel<<<(E + 255) / 256, 256, 0, stream>>>(ei, cnt, E);
  scan_kernel<<<1, 1024, 0, stream>>>(cnt, rowptr, cursor, logsum, N, E);
  scatter_kernel<<<(E + 255) / 256, 256, 0, stream>>>(ei, cursor, perm, E);
  cvt_bf16_kernel<<<((N * F0) + 255) / 256, 256, 0, stream>>>(x0, xbf, N * F0);

  const float* xf = x0;
  float* xfn = xfA;
  int F = F0;
  for (int l = 0; l < 3; ++l) {
    const float* we = (const float*)d_in[3 + l * 10 + 0];
    const float* be = (const float*)d_in[3 + l * 10 + 1];
    const float* wp = (const float*)d_in[3 + l * 10 + 2];
    const float* bp = (const float*)d_in[3 + l * 10 + 3];
    const float* wo = (const float*)d_in[3 + l * 10 + 4];
    const float* bo = (const float*)d_in[3 + l * 10 + 5];
    const float* wl = (const float*)d_in[3 + l * 10 + 6];
    const float* bl = (const float*)d_in[3 + l * 10 + 7];
    const float* g  = (const float*)d_in[3 + l * 10 + 8];
    const float* bb = (const float*)d_in[3 + l * 10 + 9];
    int Fo  = (l == 0) ? 64 : (l == 1 ? 128 : 256);
    int K13 = 13 * F;

    int tE = 2 * F * F + 32 * F + F;
    fold_edge_kernel<<<(tE + 255) / 256, 256, 0, stream>>>(we, be, wp, bp, WpTd, WpTs, WeT, bm, F);
    int tN = K13 * Fo + Fo;
    fold_node_kernel<<<(tN + 255) / 256, 256, 0, stream>>>(wo, bo, wl, bl, W2T, b2, K13, Fo);

    // edge messages: NT=2 when F==32 (only 2 col tiles), else NT=4
    if (F == 32) {
      long ew = (long)(E / 16) * ((F / 16) / 2);
      edge_msg_kernel<2><<<(unsigned)((ew + 3) / 4), 128, 0, stream>>>(ei, ea, xbf, WpTd, WpTs, WeT, bm, mbuf, E, F);
    } else {
      long ew = (long)(E / 16) * ((F / 16) / 4);
      edge_msg_kernel<4><<<(unsigned)((ew + 3) / 4), 128, 0, stream>>>(ei, ea, xbf, WpTd, WpTs, WeT, bm, mbuf, E, F);
    }

    aggregate_kernel<<<(N + 3) / 4, 128, 0, stream>>>(rowptr, perm, mbuf, xf, cnt, logsum, zbuf, N, F);

    long nw = (long)(N / 16) * ((Fo / 16) / 4);
    node_gemm_kernel<4><<<(unsigned)((nw + 3) / 4), 128, 0, stream>>>(zbuf, W2T, b2, xfn, N, K13, Fo);

    bn_stats_kernel<<<Fo, 256, 0, stream>>>(xfn, bnsum, bnss, N, Fo);
    bn_apply_kernel<<<((N * Fo) + 255) / 256, 256, 0, stream>>>(xfn, bnsum, bnss, g, bb, xbf, N, Fo);

    xf = xfn;
    xfn = (xfn == xfA) ? xfB : xfA;
    F = Fo;
  }

  classify_kernel<<<((N * C) + 255) / 256, 256, 0, stream>>>(xf, wc, bc, (float*)d_out, N, F, C);
}
